// MultiHeadAttention_26542897889811
// MI455X (gfx1250) — matrix operand order repaired and checked by reading, not compiled, not run on hardware
//
#include <hip/hip_runtime.h>

// ---------------------------------------------------------------------------
// MHA forward for MI455X (gfx1250, wave32, WMMA).
//   - q/k/v/out projections: double-buffered LDS-tiled GEMMs, f16 operands,
//     fp32 accumulation via v_wmma_f32_16x16x32_f16 (8 WMMA / K-step / wave).
//   - flash attention: scores^T = K*Q^T so the probability tile converts to
//     the P A-fragment entirely in-register; online softmax per lane; V
//     staged transposed through ping-pong LDS; mask applied as precomputed
//     additive -1e15 bias (masked probs underflow to exact 0).
// ---------------------------------------------------------------------------

typedef _Float16 half_t;
typedef __attribute__((ext_vector_type(16))) _Float16 v16h;
typedef __attribute__((ext_vector_type(8)))  _Float16 v8h;
typedef __attribute__((ext_vector_type(8)))  float    v8f;
typedef __attribute__((ext_vector_type(4)))  float    v4f;

#define D_MODEL 1024
#define N_HEADS 16
#define HEAD_DIM 64
#define SEQ 2048
#define BATCH 4
#define LDSPAD 40   // padded row stride in halves: bank-conflict-free b128 reads

static __device__ __forceinline__ v16h cat16(v8h lo, v8h hi) {
  return __builtin_shufflevector(lo, hi, 0,1,2,3,4,5,6,7,8,9,10,11,12,13,14,15);
}
static __device__ __forceinline__ v8f zero8() {
  v8f z = {0.f,0.f,0.f,0.f,0.f,0.f,0.f,0.f};
  return z;
}

// ---------------------------------------------------------------------------
// mask -> additive softmax bias (query-invariant, computed once per launch)
// ---------------------------------------------------------------------------
__global__ __launch_bounds__(256)
void mask_bias_kernel(const int* __restrict__ mask, float* __restrict__ biasb)
{
  const int i = blockIdx.x * 256 + threadIdx.x;
  biasb[i] = (mask[i] == 0) ? -1.0e15f : 0.0f;
}

// ---------------------------------------------------------------------------
// GEMM: C[M x 1024] = A[M x 1024] * W[1024 x 1024] + bias, M = BATCH*SEQ.
// Block tile 128x128, K-step 32, 8 waves, ping-pong LDS double buffering:
// global loads for step i+1 are issued before the WMMAs of step i, one
// barrier per K-step.
// ---------------------------------------------------------------------------
template<bool TO_HEADS, bool A_F16>
__global__ __launch_bounds__(256)
void gemm_bias_kernel(const void* __restrict__ Ap,
                      const float* __restrict__ W,
                      const float* __restrict__ bias,
                      half_t* __restrict__ out_h,
                      float* __restrict__ out_f,
                      float scale)
{
  __shared__ half_t As[2][128 * LDSPAD];   // [m][k]
  __shared__ half_t Bs[2][128 * LDSPAD];   // [n][k] (W transposed)

  const int tid  = threadIdx.x;
  const int lane = tid & 31;
  const int w    = tid >> 5;
  const int wm   = w & 3;          // M quadrant (32 rows each)
  const int wn   = w >> 2;         // N half (64 cols each)
  const int lm   = lane & 15;
  const int hf   = lane >> 4;
  const int bm   = blockIdx.y * 128;
  const int bn   = blockIdx.x * 128;

  // Cooperative load mapping
  const int la_row = tid >> 1;            // 0..127
  const int la_k   = (tid & 1) * 16;      // 0 or 16
  const int lb_k   = tid >> 3;            // 0..31
  const int lb_n   = (tid & 7) * 16;      // 0..112

  v8f acc[2][4];
#pragma unroll
  for (int i = 0; i < 2; ++i)
#pragma unroll
    for (int j = 0; j < 4; ++j) acc[i][j] = zero8();

  // ---- staging: global -> registers ----
  v8h ha0, ha1;     // A tile slice (16 halves)
  float wr[16];     // W tile slice (16 floats)

  auto gload = [&](int kb) {
    if constexpr (A_F16) {
      const half_t* A = (const half_t*)Ap;
      v16h av = *(const v16h*)(A + (size_t)(bm + la_row) * 1024 + kb + la_k);
      ha0 = __builtin_shufflevector(av, av, 0,1,2,3,4,5,6,7);
      ha1 = __builtin_shufflevector(av, av, 8,9,10,11,12,13,14,15);
    } else {
      const float* src = (const float*)Ap + (size_t)(bm + la_row) * 1024 + kb + la_k;
#pragma unroll
      for (int j = 0; j < 8; ++j) { ha0[j] = (half_t)src[j]; ha1[j] = (half_t)src[8 + j]; }
    }
    const float* wsrc = W + (size_t)(kb + lb_k) * 1024 + bn + lb_n;
#pragma unroll
    for (int j = 0; j < 16; ++j) wr[j] = wsrc[j];
  };
  auto stage = [&](int bsel) {
    *(v8h*)(&As[bsel][la_row * LDSPAD + la_k])     = ha0;
    *(v8h*)(&As[bsel][la_row * LDSPAD + la_k + 8]) = ha1;
#pragma unroll
    for (int j = 0; j < 16; ++j)
      Bs[bsel][(lb_n + j) * LDSPAD + lb_k] = (half_t)wr[j];
  };

  gload(0);
  stage(0);
  int buf = 0;

  for (int kb = 0; kb < 1024; kb += 32) {
    __syncthreads();                       // LDS[buf] ready / prev reads done
    const bool more = (kb + 32) < 1024;
    if (more) gload(kb + 32);              // overlap with WMMAs below

    // ---- fragments + WMMA from LDS[buf] ----
    v16h bf[4];
#pragma unroll
    for (int nt = 0; nt < 4; ++nt) {
      const half_t* p = &Bs[buf][(wn * 64 + nt * 16 + lm) * LDSPAD + hf * 8];
      bf[nt] = cat16(*(const v8h*)p, *(const v8h*)(p + 16));
    }
#pragma unroll
    for (int mt = 0; mt < 2; ++mt) {
      const half_t* p = &As[buf][(wm * 32 + mt * 16 + lm) * LDSPAD + hf * 8];
      v16h af = cat16(*(const v8h*)p, *(const v8h*)(p + 16));
#pragma unroll
      for (int nt = 0; nt < 4; ++nt)
        acc[mt][nt] = __builtin_amdgcn_wmma_f32_16x16x32_f16(
            false, af, false, bf[nt], (short)0, acc[mt][nt], false, false);
    }

    if (more) stage(buf ^ 1);              // fill idle buffer
    buf ^= 1;
  }

  // ---- epilogue ----
#pragma unroll
  for (int nt = 0; nt < 4; ++nt) {
    const int c = bn + wn * 64 + nt * 16 + lm;
    const float bv = bias[c];
#pragma unroll
    for (int mt = 0; mt < 2; ++mt) {
#pragma unroll
      for (int r = 0; r < 8; ++r) {
        const int row = bm + wm * 32 + mt * 16 + r + hf * 8;
        const float v = (acc[mt][nt][r] + bv) * scale;
        if constexpr (TO_HEADS) {
          const int b = row >> 11, s = row & 2047;
          const int h = c >> 6,  hd = c & 63;
          out_h[(((size_t)b * N_HEADS + h) * SEQ + s) * HEAD_DIM + hd] = (half_t)v;
        } else {
          out_f[(size_t)row * 1024 + c] = v;
        }
      }
    }
  }
}

// ---------------------------------------------------------------------------
// Flash attention: grid (S/64, H, B), 128 threads (4 waves x 16 queries).
// ---------------------------------------------------------------------------
__global__ __launch_bounds__(128)
void attn_kernel(const half_t* __restrict__ qbuf, const half_t* __restrict__ kbuf,
                 const half_t* __restrict__ vbuf, const float* __restrict__ biasbuf,
                 half_t* __restrict__ attn)
{
  __shared__ half_t vT[2][HEAD_DIM * LDSPAD];   // V^T: [hd][key], padded

  const int tid  = threadIdx.x;
  const int lane = tid & 31;
  const int w    = tid >> 5;
  const int lm   = lane & 15;
  const int hf   = lane >> 4;
  const int b    = blockIdx.z, h = blockIdx.y;
  const size_t bh = (size_t)b * N_HEADS + h;
  const half_t* qh = qbuf + bh * SEQ * HEAD_DIM;
  const half_t* kh = kbuf + bh * SEQ * HEAD_DIM;
  const half_t* vh = vbuf + bh * SEQ * HEAD_DIM;
  const float* mb = biasbuf + b * SEQ;
  const int q0 = blockIdx.x * 64 + w * 16;

  // Q B-fragments (N = query = lm), kept in registers for whole key loop.
  v16h qf[2];
#pragma unroll
  for (int c = 0; c < 2; ++c)
    qf[c] = *(const v16h*)(qh + (size_t)(q0 + lm) * HEAD_DIM + c * 32 + hf * 16);

  v8f ctx[4];
#pragma unroll
  for (int t = 0; t < 4; ++t) ctx[t] = zero8();
  float run_m = -3.0e38f, run_s = 0.f;

  const int vkey = tid >> 2;          // 0..31
  const int vhd0 = (tid & 3) * 16;    // 0,16,32,48

  // prologue: stage V block 0
  v16h vstage = *(const v16h*)(vh + (size_t)vkey * HEAD_DIM + vhd0);
#pragma unroll
  for (int j = 0; j < 16; ++j) vT[0][(vhd0 + j) * LDSPAD + vkey] = vstage[j];
  int buf = 0;

  for (int k0 = 0; k0 < SEQ; k0 += 32) {
    __syncthreads();                      // vT[buf] ready / prev reads done
    const bool more = (k0 + 32) < SEQ;
    if (more)                             // overlap next V load with compute
      vstage = *(const v16h*)(vh + (size_t)(k0 + 32 + vkey) * HEAD_DIM + vhd0);

    // scores^T: 2 key tiles x (2 WMMA over Hd=64)
    v8f sc[2];
#pragma unroll
    for (int t = 0; t < 2; ++t) {
      v8f cz = zero8();
#pragma unroll
      for (int c = 0; c < 2; ++c) {
        const half_t* p = kh + (size_t)(k0 + t * 16 + lm) * HEAD_DIM + c * 32 + hf * 16;
        v16h af = cat16(*(const v8h*)p, *(const v8h*)(p + 8));
        cz = __builtin_amdgcn_wmma_f32_16x16x32_f16(
            false, af, false, qf[c], (short)0, cz, false, false);
      }
      sc[t] = cz;
    }

    // additive key-padding bias (precomputed -1e15 / 0)
#pragma unroll
    for (int t = 0; t < 2; ++t) {
      const float* bp = mb + k0 + t * 16 + hf * 8;
      v4f b0 = *(const v4f*)bp;
      v4f b1 = *(const v4f*)(bp + 4);
#pragma unroll
      for (int r = 0; r < 4; ++r) {
        sc[t][r]     += b0[r];
        sc[t][r + 4] += b1[r];
      }
    }

    // online softmax (row = query = lm; merge lane halves with xor16)
    float mx = run_m;
#pragma unroll
    for (int t = 0; t < 2; ++t)
#pragma unroll
      for (int r = 0; r < 8; ++r) mx = fmaxf(mx, sc[t][r]);
    mx = fmaxf(mx, __shfl_xor(mx, 16));
    const float corr = __expf(run_m - mx);
    run_m = mx;

    float pv[16];
    float psum = 0.f;
#pragma unroll
    for (int r = 0; r < 8; ++r) {
      pv[r]     = __expf(sc[0][r] - mx);
      pv[8 + r] = __expf(sc[1][r] - mx);
      psum += pv[r] + pv[8 + r];
    }
    psum += __shfl_xor(psum, 16);
    run_s = run_s * corr + psum;

    // rescale ctx: factor for query (r + 8*hf) lives in that lane
#pragma unroll
    for (int r = 0; r < 8; ++r) {
      const float f = __shfl(corr, (hf << 3) + r);
#pragma unroll
      for (int t = 0; t < 4; ++t) ctx[t][r] *= f;
    }

    // P A-fragment entirely in-register (scores^T layout == A layout)
    v16h pf;
#pragma unroll
    for (int i = 0; i < 16; ++i) pf[i] = (half_t)pv[i];

    // ctx += P * V  (4 hd tiles)
#pragma unroll
    for (int ht = 0; ht < 4; ++ht) {
      const half_t* p = &vT[buf][(ht * 16 + lm) * LDSPAD + hf * 8];
      v16h vf = cat16(*(const v8h*)p, *(const v8h*)(p + 16));
      ctx[ht] = __builtin_amdgcn_wmma_f32_16x16x32_f16(
          false, pf, false, vf, (short)0, ctx[ht], false, false);
    }

    if (more) {                           // fill idle V buffer
#pragma unroll
      for (int j = 0; j < 16; ++j)
        vT[buf ^ 1][(vhd0 + j) * LDSPAD + vkey] = vstage[j];
    }
    buf ^= 1;
  }

  // finalize: divide by running sum, write f16 [B,S,D] for output projection
  const float inv = 1.0f / run_s;
#pragma unroll
  for (int r = 0; r < 8; ++r) {
    const float f = __shfl(inv, (hf << 3) + r);
    const int s = q0 + hf * 8 + r;
    half_t* op = attn + ((size_t)b * SEQ + s) * D_MODEL + h * HEAD_DIM;
#pragma unroll
    for (int ht = 0; ht < 4; ++ht)
      op[ht * 16 + lm] = (half_t)(ctx[ht][r] * f);
  }
}

// ---------------------------------------------------------------------------
extern "C" void kernel_launch(void* const* d_in, const int* in_sizes, int n_in,
                              void* d_out, int out_size, void* d_ws, size_t ws_size,
                              hipStream_t stream)
{
  const float* Q    = (const float*)d_in[0];
  const float* K    = (const float*)d_in[1];
  const float* V    = (const float*)d_in[2];
  const int*   mask = (const int*)  d_in[3];
  const float* Wq   = (const float*)d_in[4];
  const float* bq   = (const float*)d_in[5];
  const float* Wk   = (const float*)d_in[6];
  const float* bk   = (const float*)d_in[7];
  const float* Wv   = (const float*)d_in[8];
  const float* bv   = (const float*)d_in[9];
  const float* Wo   = (const float*)d_in[10];
  const float* bo   = (const float*)d_in[11];

  const size_t per = (size_t)BATCH * N_HEADS * SEQ * HEAD_DIM;  // 8M halves
  half_t* qbuf = (half_t*)d_ws;
  half_t* kbuf = qbuf + per;
  half_t* vbuf = kbuf + per;
  half_t* abuf = vbuf + per;                 // attention context, [B,S,D] f16
  float*  mbias = (float*)(abuf + per);      // additive mask bias, [B,S] f32

  mask_bias_kernel<<<dim3((BATCH * SEQ) / 256), 256, 0, stream>>>(mask, mbias);

  dim3 gg(D_MODEL / 128, (BATCH * SEQ) / 128, 1);   // (8, 64)
  gemm_bias_kernel<true,  false><<<gg, 256, 0, stream>>>(Q, Wq, bq, qbuf, nullptr, 0.125f);
  gemm_bias_kernel<true,  false><<<gg, 256, 0, stream>>>(K, Wk, bk, kbuf, nullptr, 1.0f);
  gemm_bias_kernel<true,  false><<<gg, 256, 0, stream>>>(V, Wv, bv, vbuf, nullptr, 1.0f);

  dim3 ga(SEQ / 64, N_HEADS, BATCH);
  attn_kernel<<<ga, 128, 0, stream>>>(qbuf, kbuf, vbuf, mbias, abuf);

  gemm_bias_kernel<false, true ><<<gg, 256, 0, stream>>>(abuf, Wo, bo, nullptr, (float*)d_out, 1.0f);
}
